// FeatureAlign_19997367730267
// MI455X (gfx1250) — matrix operand order repaired and checked by reading, not compiled, not run on hardware
//
#include <hip/hip_runtime.h>
#include <hip/hip_bf16.h>
#include <stdint.h>

typedef __attribute__((ext_vector_type(16))) _Float16 v16h;
typedef __attribute__((ext_vector_type(8)))  _Float16 v8h;
typedef __attribute__((ext_vector_type(8)))  float    v8f;

#define D_    16
#define H_    96
#define W_    96
#define CIN   16
#define HW    (H_ * W_)
#define SITES (D_ * H_ * W_)    // 147456
#define K3    27
#define KOFF  (CIN * K3)        // 432
#define KP1   448               // 432 padded to mult of 32
#define MOFF  81
#define MP1   96                // 81 padded to mult of 16
#define CEMB  32
#define KFLOW (CEMB * K3)       // 864
#define KP3   896               // 864 padded to mult of 32

// ---------------------------------------------------------------------------
// CDNA5 async global->LDS copy (ASYNCcnt path), inline asm per ISA §15.18.
// VDST holds the wave-relative LDS byte offset; HW adds LDS_BASE.
// The low 32 bits of a generic pointer to a __shared__ object are exactly
// that DS-domain offset on AMDGPU.
// ---------------------------------------------------------------------------
__device__ static inline void async_copy_b32(const float* gptr, void* lptr) {
  unsigned long long ga = (unsigned long long)(uintptr_t)gptr;
  unsigned lo = (unsigned)(uintptr_t)lptr;
  asm volatile("global_load_async_to_lds_b32 %0, %1, off"
               :: "v"(lo), "v"(ga) : "memory");
}
__device__ static inline void async_wait0() {
  asm volatile("s_wait_asynccnt 0x0" ::: "memory");
}

// ---------------------------------------------------------------------------
// WMMA helpers (CDNA5 wave32, 16x16x32 f16 -> f32)
// ---------------------------------------------------------------------------
__device__ static inline v8f wmma16(v16h a, v16h b, v8f c) {
  // 8 args: (neg_a, A, neg_b, B, c_mod, C, reuse_a, reuse_b)
  return __builtin_amdgcn_wmma_f32_16x16x32_f16(
      false, a, false, b, (short)0, c, false, false);
}

// A matrix stored row-major [Mpad][Kpad] in f16.
// ISA 16-bit A 16x32 layout: lane<16 holds row M=lane, K in {0..7} U {16..23};
// lane>=16 holds row M=lane-16, K shifted by +8. Each half is 8 contiguous f16
// = one 16-byte load, so a fragment is exactly two v8h loads.
__device__ static inline v16h load_frag_a(const _Float16* __restrict__ A,
                                          int Kpad, int mt, int ks, int lane) {
  int half = lane >> 4;
  int m = lane & 15;
  const _Float16* row = A + (size_t)(mt * 16 + m) * Kpad;
  union { v16h v; v8h h[2]; } u;
  u.h[0] = *(const v8h*)(row + ks + half * 8);
  u.h[1] = *(const v8h*)(row + ks + 16 + half * 8);
  return u.v;
}

// B tile stored col-major [n][Kpad] in LDS.
// 16-bit B 32x16 layout (matches A): lane<16 -> column N=lane, K in {0..7} U {16..23};
// lane>=16 -> column N=lane-16, K shifted by +8 -> two 16-byte loads per lane.
__device__ static inline v16h load_frag_b(const _Float16* __restrict__ Bt,
                                          int Kpad, int ks, int lane) {
  int half = lane >> 4;
  int n = lane & 15;
  const _Float16* col = Bt + (size_t)n * Kpad + ks;
  union { v16h v; v8h h[2]; } u;
  u.h[0] = *(const v8h*)(col + half * 8);
  u.h[1] = *(const v8h*)(col + 16 + half * 8);
  return u.v;
}

// ---------------------------------------------------------------------------
// Kernel 0: repack conv weights into padded f16 A matrices
// ---------------------------------------------------------------------------
__global__ __launch_bounds__(256) void prep_weights(
    const float* __restrict__ w_off, const float* __restrict__ w_def,
    const float* __restrict__ w_out, _Float16* __restrict__ A_off,
    _Float16* __restrict__ A_def, _Float16* __restrict__ A_out) {
  int i = blockIdx.x * 256 + threadIdx.x;
  if (i < MP1 * KP1) {
    int m = i / KP1, k = i - m * KP1;
    A_off[i] = (_Float16)((m < MOFF && k < KOFF) ? w_off[m * KOFF + k] : 0.0f);
  }
  if (i < CEMB * KP1) {
    int m = i / KP1, k = i - m * KP1;
    A_def[i] = (_Float16)((m < CEMB && k < KOFF) ? w_def[m * KOFF + k] : 0.0f);
  }
  if (i < 16 * KP3) {
    int m = i / KP3, k = i - m * KP3;
    A_out[i] = (_Float16)((m < 2 && k < KFLOW) ? w_out[m * KFLOW + k] : 0.0f);
  }
}

// ---------------------------------------------------------------------------
// Kernel 1: fused offset-conv (WMMA) -> trilinear deform sampling ->
//           deform einsum (WMMA) -> feat[32][SITES]
// One wave per 16-site tile (16 consecutive x positions in one row).
// The 3x3x18 halo of x is staged into LDS with async global->LDS copies.
// ---------------------------------------------------------------------------
__global__ __launch_bounds__(32) void k_deform(
    const float* __restrict__ x, const float* __restrict__ b_off,
    const _Float16* __restrict__ A_off, const _Float16* __restrict__ A_def,
    float* __restrict__ feat) {
  __shared__ _Float16 sB[16 * KP1];     // im2col tile, col-major [n][KP1]
  __shared__ float sOff[MP1 * 16];      // 81 offset channels x 16 sites
  __shared__ float sX[CIN * 9 * 18];    // halo: [c][kd3*3+kh3][i], i = x0-1..x0+16

  const int lane = threadIdx.x;
  const int half = lane >> 4;
  const int nn = lane & 15;
  const int siteBase = blockIdx.x * 16;
  const int z = siteBase / HW;
  const int rem = siteBase - z * HW;
  const int y = rem / W_;
  const int x0w = rem - y * W_;

  // ---- phase 0: async-stage the x halo into LDS ----
  const int NST1 = CIN * 9 * 18;  // 2592
  for (int t = lane; t < NST1; t += 32) {
    int i = t % 18;
    int r = t / 18;            // r = c*9 + kd3*3 + kh3
    int c = r / 9;
    int kd3 = (r % 9) / 3, kh3 = (r % 9) % 3;
    int zz = z + kd3 - 1, yy = y + kh3 - 1, xx = x0w - 1 + i;
    bool valid = zz >= 0 && zz < D_ && yy >= 0 && yy < H_ && xx >= 0 && xx < W_;
    if (valid) {
      async_copy_b32(&x[c * SITES + (zz * H_ + yy) * W_ + xx], &sX[t]);
    } else {
      sX[t] = 0.0f;  // disjoint from async targets -> no race
    }
  }
  async_wait0();
  __syncthreads();

  // ---- phase 1: im2col from staged halo, f32 -> f16 ----
  for (int idx = lane; idx < 16 * KP1; idx += 32) {
    int n = idx / KP1, k = idx - n * KP1;
    float v = 0.0f;
    if (k < KOFF) {
      int c = k / K3, tap = k - c * K3;
      int kd3 = tap / 9, kh3 = (tap / 3) % 3, kw3 = tap % 3;
      v = sX[(c * 9 + kd3 * 3 + kh3) * 18 + (n + kw3)];
    }
    sB[n * KP1 + k] = (_Float16)v;
  }
  __syncthreads();

  // ---- phase 2: offset GEMM, 6 M-tiles of 16 (81 padded to 96) ----
  for (int mt = 0; mt < 6; ++mt) {
    v8f acc = {};
    for (int ks = 0; ks < KP1; ks += 32) {
      v16h a = load_frag_a(A_off, KP1, mt, ks, lane);
      v16h b = load_frag_b(sB, KP1, ks, lane);
      acc = wmma16(a, b, acc);
    }
#pragma unroll
    for (int r = 0; r < 8; ++r) {
      int m = mt * 16 + r + 8 * half;
      if (m < MOFF) sOff[m * 16 + nn] = acc[r] + b_off[m];
    }
  }
  __syncthreads();

  // ---- phase 3: trilinear sampling -> new im2col tile in sB ----
  for (int idx = lane; idx < 16 * (KP1 - KOFF); idx += 32) {
    int n = idx >> 4, k = KOFF + (idx & 15);
    sB[n * KP1 + k] = (_Float16)0.0f;  // K pad
  }
  for (int q = lane; q < 16 * K3; q += 32) {
    int n = q & 15, tap = q >> 4;
    int kd = tap / 9 - 1, kh = (tap / 3) % 3 - 1, kw = tap % 3 - 1;
    float zf = (float)(z + kd) + sOff[(0 * K3 + tap) * 16 + n];
    float yf = (float)(y + kh) + sOff[(1 * K3 + tap) * 16 + n];
    float xf = (float)(x0w + n + kw) + sOff[(2 * K3 + tap) * 16 + n];
    float z0 = floorf(zf), y0 = floorf(yf), x0 = floorf(xf);
    float accv[CIN];
#pragma unroll
    for (int c = 0; c < CIN; ++c) accv[c] = 0.0f;
#pragma unroll
    for (int corner = 0; corner < 8; ++corner) {
      float zi = z0 + (float)(corner >> 2);
      float yi = y0 + (float)((corner >> 1) & 1);
      float xi = x0 + (float)(corner & 1);
      float wgt = (1.0f - fabsf(zf - zi)) * (1.0f - fabsf(yf - yi)) *
                  (1.0f - fabsf(xf - xi));
      bool valid = (zi >= 0.0f) && (zi < (float)D_) && (yi >= 0.0f) &&
                   (yi < (float)H_) && (xi >= 0.0f) && (xi < (float)W_);
      if (valid) {
        int gi = ((int)zi * H_ + (int)yi) * W_ + (int)xi;
#pragma unroll
        for (int c = 0; c < CIN; ++c) accv[c] += wgt * x[c * SITES + gi];
      }
    }
#pragma unroll
    for (int c = 0; c < CIN; ++c)
      sB[n * KP1 + c * K3 + tap] = (_Float16)accv[c];
  }
  __syncthreads();

  // ---- phase 4: deform GEMM, 2 M-tiles -> feat ----
  for (int mt = 0; mt < 2; ++mt) {
    v8f acc = {};
    for (int ks = 0; ks < KP1; ks += 32) {
      v16h a = load_frag_a(A_def, KP1, mt, ks, lane);
      v16h b = load_frag_b(sB, KP1, ks, lane);
      acc = wmma16(a, b, acc);
    }
#pragma unroll
    for (int r = 0; r < 8; ++r) {
      int m = mt * 16 + r + 8 * half;
      feat[(size_t)m * SITES + siteBase + nn] = acc[r];
    }
  }
}

// ---------------------------------------------------------------------------
// Kernel 2: batch-norm stats (deterministic per-channel tree reduction)
// -> fused scale/shift
// ---------------------------------------------------------------------------
__global__ __launch_bounds__(256) void k_bnstats(
    const float* __restrict__ feat, const float* __restrict__ gamma,
    const float* __restrict__ beta, float* __restrict__ scale,
    float* __restrict__ shift) {
  __shared__ float s1[256], s2[256];
  int m = blockIdx.x;
  const float* f = feat + (size_t)m * SITES;
  float a = 0.0f, b = 0.0f;
  for (int i = threadIdx.x; i < SITES; i += 256) {
    float v = f[i];
    a += v;
    b += v * v;
  }
  s1[threadIdx.x] = a;
  s2[threadIdx.x] = b;
  __syncthreads();
  for (int s = 128; s > 0; s >>= 1) {
    if (threadIdx.x < s) {
      s1[threadIdx.x] += s1[threadIdx.x + s];
      s2[threadIdx.x] += s2[threadIdx.x + s];
    }
    __syncthreads();
  }
  if (threadIdx.x == 0) {
    float mu = s1[0] / (float)SITES;
    float var = s2[0] / (float)SITES - mu * mu;
    float rs = rsqrtf(var + 1e-5f);
    float sc = gamma[m] * rs;
    scale[m] = sc;
    shift[m] = beta[m] - mu * sc;
  }
}

// ---------------------------------------------------------------------------
// Kernel 3: flow head. The feat halo is async-staged raw into LDS; the
// im2col applies scale/shift/ReLU and decides zero padding from indices
// (padding is applied AFTER normalization in the reference).
// ---------------------------------------------------------------------------
__global__ __launch_bounds__(32) void k_flow(
    const float* __restrict__ feat, const float* __restrict__ scale,
    const float* __restrict__ shift, const _Float16* __restrict__ A_out,
    const float* __restrict__ b_out, float* __restrict__ flow) {
  __shared__ _Float16 sB[16 * KP3];
  __shared__ float sF[CEMB * 9 * 18];  // 5184 floats
  __shared__ float sScale[CEMB], sShift[CEMB];

  const int lane = threadIdx.x;
  const int half = lane >> 4;
  const int nn = lane & 15;
  const int siteBase = blockIdx.x * 16;
  const int z = siteBase / HW;
  const int rem = siteBase - z * HW;
  const int y = rem / W_;
  const int x0w = rem - y * W_;

  if (lane < CEMB) {
    sScale[lane] = scale[lane];
    sShift[lane] = shift[lane];
  }

  // ---- async-stage feat halo ----
  const int NST3 = CEMB * 9 * 18;  // 5184
  for (int t = lane; t < NST3; t += 32) {
    int i = t % 18;
    int r = t / 18;
    int c = r / 9;
    int kd3 = (r % 9) / 3, kh3 = (r % 9) % 3;
    int zz = z + kd3 - 1, yy = y + kh3 - 1, xx = x0w - 1 + i;
    bool valid = zz >= 0 && zz < D_ && yy >= 0 && yy < H_ && xx >= 0 && xx < W_;
    if (valid) {
      async_copy_b32(&feat[(size_t)c * SITES + (zz * H_ + yy) * W_ + xx],
                     &sF[t]);
    } else {
      sF[t] = 0.0f;
    }
  }
  async_wait0();
  __syncthreads();

  // ---- im2col with BN + ReLU fused ----
  for (int idx = lane; idx < 16 * KP3; idx += 32) {
    int n = idx / KP3, k = idx - n * KP3;
    float v = 0.0f;
    if (k < KFLOW) {
      int c = k / K3, tap = k - c * K3;
      int kd3 = tap / 9, kh3 = (tap / 3) % 3, kw3 = tap % 3;
      int zz = z + kd3 - 1, yy = y + kh3 - 1, xx = x0w + n + kw3 - 1;
      if (zz >= 0 && zz < D_ && yy >= 0 && yy < H_ && xx >= 0 && xx < W_) {
        float fv = sF[(c * 9 + kd3 * 3 + kh3) * 18 + (n + kw3)];
        fv = fv * sScale[c] + sShift[c];
        v = fv > 0.0f ? fv : 0.0f;
      }
    }
    sB[n * KP3 + k] = (_Float16)v;
  }
  __syncthreads();

  v8f acc = {};
  for (int ks = 0; ks < KP3; ks += 32) {
    v16h a = load_frag_a(A_out, KP3, 0, ks, lane);
    v16h b = load_frag_b(sB, KP3, ks, lane);
    acc = wmma16(a, b, acc);
  }
#pragma unroll
  for (int r = 0; r < 8; ++r) {
    int m = r + 8 * half;
    if (m < 2) flow[(size_t)m * SITES + siteBase + nn] = acc[r] + b_out[m];
  }
}

// ---------------------------------------------------------------------------
// Kernel 4: TiltWarp bilinear gather (zero padding)
// out[b=0][i(Cin)][d(D)][h][w] samples x[0,i,d,:,:] at
// (h + flow_y[z=i,h,w], w + flow_x[z=i,h,w]).
// ---------------------------------------------------------------------------
__global__ __launch_bounds__(256) void k_warp(const float* __restrict__ x,
                                              const float* __restrict__ flow,
                                              float* __restrict__ out) {
  int idx = blockIdx.x * 256 + threadIdx.x;
  if (idx >= CIN * D_ * H_ * W_) return;
  int w = idx % W_;
  int h = (idx / W_) % H_;
  int d = (idx / HW) % D_;
  int i = idx / (D_ * HW);
  int fsite = i * HW + h * W_ + w;
  float px = (float)w + flow[0 * SITES + fsite];
  float py = (float)h + flow[1 * SITES + fsite];
  float px0 = floorf(px), py0 = floorf(py);
  const float* img = x + (size_t)(i * D_ + d) * HW;
  float acc = 0.0f;
#pragma unroll
  for (int corner = 0; corner < 4; ++corner) {
    float yi = py0 + (float)(corner >> 1);
    float xi = px0 + (float)(corner & 1);
    float wgt = (1.0f - fabsf(py - yi)) * (1.0f - fabsf(px - xi));
    if (yi >= 0.0f && yi < (float)H_ && xi >= 0.0f && xi < (float)W_)
      acc += wgt * img[(int)yi * W_ + (int)xi];
  }
  out[idx] = acc;
}

// ---------------------------------------------------------------------------
// Host-side launch
// ---------------------------------------------------------------------------
extern "C" void kernel_launch(void* const* d_in, const int* in_sizes, int n_in,
                              void* d_out, int out_size, void* d_ws,
                              size_t ws_size, hipStream_t stream) {
  (void)in_sizes; (void)n_in; (void)out_size; (void)ws_size;
  const float* x     = (const float*)d_in[0];
  const float* w_off = (const float*)d_in[1];
  const float* b_off = (const float*)d_in[2];
  const float* w_def = (const float*)d_in[3];
  const float* gamma = (const float*)d_in[4];
  const float* beta  = (const float*)d_in[5];
  const float* w_out = (const float*)d_in[6];
  const float* b_out = (const float*)d_in[7];
  float* out = (float*)d_out;

  char* ws = (char*)d_ws;
  size_t o = 0;
  auto wsalloc = [&](size_t bytes) {
    char* p = ws + o;
    o += (bytes + 255) & ~(size_t)255;
    return p;
  };
  float*    feat  = (float*)wsalloc(sizeof(float) * CEMB * SITES);    // 18.9 MB
  float*    flow  = (float*)wsalloc(sizeof(float) * 2 * SITES);       //  1.2 MB
  float*    scale = (float*)wsalloc(sizeof(float) * CEMB);
  float*    shift = (float*)wsalloc(sizeof(float) * CEMB);
  _Float16* A_off = (_Float16*)wsalloc(sizeof(_Float16) * MP1 * KP1);
  _Float16* A_def = (_Float16*)wsalloc(sizeof(_Float16) * CEMB * KP1);
  _Float16* A_out = (_Float16*)wsalloc(sizeof(_Float16) * 16 * KP3);

  prep_weights<<<(MP1 * KP1 + 255) / 256, 256, 0, stream>>>(
      w_off, w_def, w_out, A_off, A_def, A_out);

  int tiles = SITES / 16;  // 9216 one-wave tiles
  k_deform<<<tiles, 32, 0, stream>>>(x, b_off, A_off, A_def, feat);
  k_bnstats<<<CEMB, 256, 0, stream>>>(feat, gamma, beta, scale, shift);
  k_flow<<<tiles, 32, 0, stream>>>(feat, scale, shift, A_out, b_out, flow);

  int N = CIN * D_ * H_ * W_;
  k_warp<<<(N + 255) / 256, 256, 0, stream>>>(x, flow, out);
}
